// StackedUnidirLSTMDecoder_23931557773550
// MI455X (gfx1250) — matrix operand order repaired and checked by reading, not compiled, not run on hardware
//
#include <hip/hip_runtime.h>
#include <hip/hip_bf16.h>

typedef __bf16 bf16_t;
typedef bf16_t v16bf __attribute__((ext_vector_type(16)));
typedef bf16_t v8bf  __attribute__((ext_vector_type(8)));
typedef float  v8f   __attribute__((ext_vector_type(8)));

#define LAYERS 4
#define BATCH  64
#define HID    512
#define TSTEPS 256
#define KDIM   1024   // fused K = 2*H  ([inp | h])
#define NG     2048   // 4*H gate width
#define NWG    32     // workgroups in persistent grid
#define TPB    128    // 4 waves/WG -> 128 waves total = 4 Mtiles * 32 Jtiles

union V16 { v16bf v; v8bf h[2]; };

// ---------------- device-scope split barrier (persistent grid) ----------------
__device__ __forceinline__ void grid_barrier(unsigned* cnt, unsigned* gen) {
  __threadfence();
  __syncthreads();
  if (threadIdx.x == 0) {
    unsigned g = __hip_atomic_load(gen, __ATOMIC_RELAXED, __HIP_MEMORY_SCOPE_AGENT);
    unsigned prev = __hip_atomic_fetch_add(cnt, 1u, __ATOMIC_ACQ_REL, __HIP_MEMORY_SCOPE_AGENT);
    if (prev == NWG - 1u) {
      __hip_atomic_store(cnt, 0u, __ATOMIC_RELAXED, __HIP_MEMORY_SCOPE_AGENT);
      __hip_atomic_fetch_add(gen, 1u, __ATOMIC_ACQ_REL, __HIP_MEMORY_SCOPE_AGENT);
    } else {
      while (__hip_atomic_load(gen, __ATOMIC_ACQUIRE, __HIP_MEMORY_SCOPE_AGENT) == g) {
        __builtin_amdgcn_s_sleep(2);
      }
    }
  }
  __syncthreads();
}

// ---- branch-free activations on TRANS ops (v_exp_f32 / v_rcp_f32), no exec-mask paths ----
__device__ __forceinline__ float fast_sigmoid(float x) {
  return __builtin_amdgcn_rcpf(1.0f + __expf(-x));
}
__device__ __forceinline__ float fast_tanh(float x) {
  // tanh(x) = 1 - 2/(exp(2x)+1); exp->inf/0 gives exact +-1 saturation
  return 1.0f - 2.0f * __builtin_amdgcn_rcpf(1.0f + __expf(2.0f * x));
}

// ------------- one-time fp32 -> bf16 weight repack: wbf[l][n][k], k = [Wih | Whh] -------------
__global__ void cvt_weights(const float* __restrict__ Wih, const float* __restrict__ Whh,
                            bf16_t* __restrict__ wbf) {
  int idx = blockIdx.x * 256 + threadIdx.x;        // over LAYERS*NG*KDIM = 8388608
  int k  = idx & (KDIM - 1);
  int nl = idx >> 10;                              // l*NG + n
  float v = (k < HID) ? Wih[nl * HID + k] : Whh[nl * HID + (k - HID)];
  wbf[idx] = (bf16_t)v;
}

// ------------- init bias sum, c state, activation double-buffers, barrier -------------
__global__ void init_state(const float* __restrict__ y, const float* __restrict__ h0,
                           const float* __restrict__ c0, const float* __restrict__ bih,
                           const float* __restrict__ bhh, float* __restrict__ bias,
                           float* __restrict__ cst, bf16_t* __restrict__ xh,
                           unsigned* __restrict__ bar) {
  int idx = blockIdx.x * 256 + threadIdx.x;        // over LAYERS*BATCH*HID = 131072
  if (idx < LAYERS * NG) bias[idx] = bih[idx] + bhh[idx];
  cst[idx] = c0[idx];
  int l = idx >> 15;          // /(BATCH*HID)
  int m = (idx >> 9) & (BATCH - 1);
  int j = idx & (HID - 1);
  bf16_t hb = (bf16_t)h0[idx];
  // xh[parity=0][l][m][HID + j] = h0
  xh[((size_t)(0 * LAYERS + l) * BATCH + m) * KDIM + HID + j] = hb;
  // xh[0][0][m][j] = y (step-0 input to layer 0)
  if (l == 0) xh[((size_t)0 * BATCH + m) * KDIM + j] = (bf16_t)y[m * HID + j];
  if (idx < 2) bar[idx] = 0u;
}

// ---------------- persistent fused LSTM decoder ----------------
__global__ __launch_bounds__(TPB, 1) void lstm_persistent(
    const bf16_t* __restrict__ wbf, const float* __restrict__ bias,
    bf16_t* __restrict__ xh, float* __restrict__ cst,
    float* __restrict__ out, unsigned* __restrict__ bar) {
  const int lane  = threadIdx.x & 31;
  const int wid   = blockIdx.x * (TPB / 32) + (threadIdx.x >> 5);
  const int mt    = wid >> 5;          // 0..3   (M tile of 64)
  const int jt    = wid & 31;          // 0..31  (j tile of 512)
  const int half  = lane >> 4;         // lane group 0/1
  const int l15   = lane & 15;

  const int mrow = mt * 16 + l15;      // A row this lane supplies
  const int ka0  = half * 16;          // A: lanes 0-15 K=0..15, lanes 16-31 K=16..31
  const int kb0  = half * 16;          // B: lanes 0-15 K=0..15, lanes 16-31 K=16..31
  const int nrow = jt * 16 + l15;      // weight row (within gate block)

  for (int t = 0; t < TSTEPS; ++t) {
    const int p = t & 1;
    for (int l = 0; l < LAYERS; ++l) {
      const bf16_t* Abase = xh + ((size_t)(p * LAYERS + l) * BATCH + mrow) * KDIM;
      const bf16_t* Wl    = wbf + (size_t)l * NG * KDIM;
      const bf16_t* Wg0   = Wl + ((size_t)(0 * HID + nrow)) * KDIM + kb0;
      const bf16_t* Wg1   = Wl + ((size_t)(1 * HID + nrow)) * KDIM + kb0;
      const bf16_t* Wg2   = Wl + ((size_t)(2 * HID + nrow)) * KDIM + kb0;
      const bf16_t* Wg3   = Wl + ((size_t)(3 * HID + nrow)) * KDIM + kb0;

      v8f acc0 = {}, acc1 = {}, acc2 = {}, acc3 = {};

      #pragma unroll 2
      for (int kk = 0; kk < KDIM; kk += 32) {
        V16 a;
        a.h[0] = *(const v8bf*)(Abase + kk + ka0);        // K = kk + half*16 .. +7
        a.h[1] = *(const v8bf*)(Abase + kk + ka0 + 8);    // K = +8 .. +15
        v16bf b0 = *(const v16bf*)(Wg0 + kk);             // 32 contiguous bytes
        v16bf b1 = *(const v16bf*)(Wg1 + kk);
        v16bf b2 = *(const v16bf*)(Wg2 + kk);
        v16bf b3 = *(const v16bf*)(Wg3 + kk);
        acc0 = __builtin_amdgcn_wmma_f32_16x16x32_bf16(false, a.v, false, b0, (short)0, acc0, false, false);
        acc1 = __builtin_amdgcn_wmma_f32_16x16x32_bf16(false, a.v, false, b1, (short)0, acc1, false, false);
        acc2 = __builtin_amdgcn_wmma_f32_16x16x32_bf16(false, a.v, false, b2, (short)0, acc2, false, false);
        acc3 = __builtin_amdgcn_wmma_f32_16x16x32_bf16(false, a.v, false, b3, (short)0, acc3, false, false);
      }

      // ---- fused LSTM cell on the 16x16 (m,j) patch held in registers ----
      const int j = jt * 16 + l15;                       // D: n = lane%16
      const float bi = bias[l * NG + 0 * HID + j];
      const float bf = bias[l * NG + 1 * HID + j];
      const float bg = bias[l * NG + 2 * HID + j];
      const float bo = bias[l * NG + 3 * HID + j];
      const int mbase = mt * 16 + half * 8;              // D: m = r + 8*(lane>=16)
      float* crow = cst + (size_t)l * BATCH * HID + j;
      // h_new destinations: (a) next-step h slot of this layer, (b) input slot downstream
      bf16_t* hdst0 = xh + (size_t)((1 - p) * LAYERS + l) * BATCH * KDIM + HID + j;
      bf16_t* hdst1 = (l < LAYERS - 1)
          ? xh + (size_t)(p * LAYERS + (l + 1)) * BATCH * KDIM + j          // feeds layer l+1 now
          : xh + (size_t)((1 - p) * LAYERS + 0) * BATCH * KDIM + j;         // feeds layer 0 next step

      #pragma unroll
      for (int r = 0; r < 8; ++r) {
        const int m = mbase + r;
        float si = fast_sigmoid(acc0[r] + bi);
        float sf = fast_sigmoid(acc1[r] + bf);
        float tg = fast_tanh(acc2[r] + bg);
        float so = fast_sigmoid(acc3[r] + bo);
        float cold = crow[(size_t)m * HID];
        float cn = sf * cold + si * tg;
        float hn = so * fast_tanh(cn);
        crow[(size_t)m * HID] = cn;
        bf16_t hb = (bf16_t)hn;
        hdst0[(size_t)m * KDIM] = hb;
        hdst1[(size_t)m * KDIM] = hb;
        if (l == LAYERS - 1) out[((size_t)m * TSTEPS + t) * HID + j] = hn;
      }

      grid_barrier(bar, bar + 1);   // publish h_new before next phase consumes it
    }
  }
}

extern "C" void kernel_launch(void* const* d_in, const int* in_sizes, int n_in,
                              void* d_out, int out_size, void* d_ws, size_t ws_size,
                              hipStream_t stream) {
  (void)in_sizes; (void)n_in; (void)out_size; (void)ws_size;
  const float* y   = (const float*)d_in[0];
  const float* h0  = (const float*)d_in[1];
  const float* c0  = (const float*)d_in[2];
  const float* Wih = (const float*)d_in[3];
  const float* Whh = (const float*)d_in[4];
  const float* bih = (const float*)d_in[5];
  const float* bhh = (const float*)d_in[6];
  float* out = (float*)d_out;

  char* ws = (char*)d_ws;
  bf16_t*   wbf  = (bf16_t*)(ws);                 // 4*2048*1024*2  = 16 MB
  float*    bias = (float*)(ws + 16777216);       // 4*2048*4       = 32 KB
  bf16_t*   xh   = (bf16_t*)(ws + 16809984);      // 2*4*64*1024*2  = 1 MB
  float*    cst  = (float*)(ws + 17858560);       // 4*64*512*4     = 512 KB
  unsigned* bar  = (unsigned*)(ws + 18382848);    // 8 B

  cvt_weights<<<dim3((LAYERS * NG * KDIM) / 256), dim3(256), 0, stream>>>(Wih, Whh, wbf);
  init_state<<<dim3((LAYERS * BATCH * HID) / 256), dim3(256), 0, stream>>>(
      y, h0, c0, bih, bhh, bias, cst, xh, bar);
  lstm_persistent<<<dim3(NWG), dim3(TPB), 0, stream>>>(wbf, bias, xh, cst, out, bar);
}
